// SparseMOE_70480413327864
// MI455X (gfx1250) — compile-verified, matrix operand order checked by reading, not run on hardware
//
#include <hip/hip_runtime.h>

// ---------------------------------------------------------------------------
// Sparse MoE (top-2) for gfx1250 / MI455X.
// B=2, S=2048 -> N=4096 tokens, D=1024, DFF=2048, E=8, TOPK=2.
// Routed compute (103 GFLOP bf16) instead of dense (413 GFLOP); fp32 weights
// stream through the 192MB L2; WMMA path: v_wmma_f32_16x16x32_bf16.
// Phase-B A-tiles staged with GLOBAL_LOAD_ASYNC_TO_LDS_B128 (ASYNCcnt).
// Workspace use: ~33.7 MB.
// ---------------------------------------------------------------------------

#define NTOK   4096
#define DMODEL 1024
#define DFFD   2048
#define NEXP   8

#ifndef __has_builtin
#define __has_builtin(x) 0
#endif
#if __has_builtin(__builtin_amdgcn_global_load_async_to_lds_b128)
#define HAVE_ASYNC_LDS 1
#else
#define HAVE_ASYNC_LDS 0
#endif

#define AS1 __attribute__((address_space(1)))
#define AS3 __attribute__((address_space(3)))

typedef int v4i __attribute__((vector_size(16)));   // builtin's pointee type

typedef __attribute__((ext_vector_type(16))) __bf16 v16bf;
typedef __attribute__((ext_vector_type(8)))  float  v8f;

union FragB { v16bf v; unsigned u[8]; };

__device__ __forceinline__ unsigned short f2bf(float f) {
  unsigned u = __float_as_uint(f);
  unsigned r = u + 0x7FFFu + ((u >> 16) & 1u);   // round-to-nearest-even
  return (unsigned short)(r >> 16);
}

__device__ __forceinline__ void wait_async_lds() {
#if HAVE_ASYNC_LDS
#if __has_builtin(__builtin_amdgcn_s_wait_asynccnt)
  __builtin_amdgcn_s_wait_asynccnt(0);
#else
  asm volatile("s_wait_asynccnt 0" ::: "memory");
#endif
#endif
}

// ------------------------------ zero init ----------------------------------
__global__ void moe_zero(float* __restrict__ outp, int* __restrict__ counts) {
  size_t i = (size_t)blockIdx.x * 256 + threadIdx.x;
  if (i < (size_t)NTOK * DMODEL) outp[i] = 0.0f;
  if (i < NEXP) counts[i] = 0;
}

// ------------------------------ router -------------------------------------
// One wave32 per token: lanes split D, shuffle-reduce 8 expert logits.
__global__ __launch_bounds__(256) void moe_router(
    const float* __restrict__ x, const float* __restrict__ Wg,
    const float* __restrict__ bg, float* __restrict__ logits,
    int* __restrict__ sel, float* __restrict__ selw, int* __restrict__ counts) {
  const int wave = threadIdx.x >> 5, lane = threadIdx.x & 31;
  const int t = blockIdx.x * 8 + wave;
  float acc[NEXP];
#pragma unroll
  for (int e = 0; e < NEXP; ++e) acc[e] = 0.f;
  const float* xr = x + (size_t)t * DMODEL;
  for (int d = lane; d < DMODEL; d += 32) {
    float xv = xr[d];
#pragma unroll
    for (int e = 0; e < NEXP; ++e) acc[e] += xv * Wg[d * NEXP + e];
  }
#pragma unroll
  for (int off = 16; off > 0; off >>= 1) {
#pragma unroll
    for (int e = 0; e < NEXP; ++e) acc[e] += __shfl_xor(acc[e], off);
  }
  if (lane == 0) {
    float lg[NEXP];
#pragma unroll
    for (int e = 0; e < NEXP; ++e) {
      lg[e] = acc[e] + bg[e];
      logits[t * NEXP + e] = lg[e];
    }
    float mx = lg[0];
#pragma unroll
    for (int e = 1; e < NEXP; ++e) mx = fmaxf(mx, lg[e]);
    float p[NEXP];
#pragma unroll
    for (int e = 0; e < NEXP; ++e) p[e] = __expf(lg[e] - mx);
    int i0 = 0;
#pragma unroll
    for (int e = 1; e < NEXP; ++e) if (p[e] > p[i0]) i0 = e;
    int i1 = (i0 == 0) ? 1 : 0;
#pragma unroll
    for (int e = 0; e < NEXP; ++e) if (e != i0 && p[e] > p[i1]) i1 = e;
    float w0 = p[i0] / (p[i0] + p[i1]);   // softmax denom cancels in rw renorm
    float w1 = p[i1] / (p[i0] + p[i1]);
    sel[t * 2 + 0] = i0;  sel[t * 2 + 1] = i1;
    selw[t * 2 + 0] = w0; selw[t * 2 + 1] = w1;
    atomicAdd(&counts[i0], 1);
    atomicAdd(&counts[i1], 1);
  }
}

// ------------------------------ scan ---------------------------------------
__global__ void moe_scan(const int* __restrict__ counts, int* __restrict__ offsets) {
  if (threadIdx.x == 0 && blockIdx.x == 0) {
    int a = 0;
    for (int e = 0; e < NEXP; ++e) { offsets[e] = a; a += counts[e]; }
    offsets[NEXP] = a;
  }
}

// --------------------- deterministic bucket placement ----------------------
// One wave per expert; ordered append via ballot prefix (wave32).
__global__ void moe_place(const int* __restrict__ sel, const float* __restrict__ selw,
                          const int* __restrict__ offsets,
                          int* __restrict__ tok_ids, float* __restrict__ tok_w) {
  const int e = blockIdx.x;
  const int lane = threadIdx.x;
  int cursor = offsets[e];
  for (int t0 = 0; t0 < NTOK; t0 += 32) {
    int t = t0 + lane;
    int s0 = sel[t * 2], s1 = sel[t * 2 + 1];
    bool flag = (s0 == e) || (s1 == e);
    float w = (s0 == e) ? selw[t * 2] : selw[t * 2 + 1];
    unsigned mask = (unsigned)__ballot(flag);
    int pos = cursor + __popc(mask & ((1u << lane) - 1u));
    if (flag) { tok_ids[pos] = t; tok_w[pos] = w; }
    cursor += __popc(mask);
  }
}

// --------------------- expert up-projection + SwiGLU -----------------------
// Block tile: 32 rows x 128 DFF cols, 8 waves. Wave w owns f-subtile w and
// both m-subtiles -> 4 WMMAs per 32-K chunk, B-fragments reused across M.
#define KSTR 40   // A-tile LDS row stride (bf16): 80B rows, conflict-free
#define WSTR 34   // B-tile LDS row stride (bf16): stride 17 dwords, coprime 64

__global__ __launch_bounds__(256) void moe_expert_up(
    const float* __restrict__ x, const float* __restrict__ Ww,
    const float* __restrict__ bw, const float* __restrict__ Wv,
    const float* __restrict__ bv, const int* __restrict__ tok_ids,
    const int* __restrict__ offsets, unsigned short* __restrict__ h) {
  __shared__ unsigned short xT[32 * KSTR];      // gathered token chunk, bf16
  __shared__ unsigned short wwT[128 * WSTR];    // Ww chunk, transposed [f][k]
  __shared__ unsigned short wvT[128 * WSTR];    // Wv chunk, transposed [f][k]
  __shared__ int stok[32];

  const int ft = blockIdx.x;             // DFF/128 tiles
  const int mt = blockIdx.y;             // 32-row tiles within expert
  const int e  = blockIdx.z;
  const int rowlo = offsets[e], rowhi = offsets[e + 1];
  const int rowbase = rowlo + mt * 32;
  if (rowbase >= rowhi) return;

  const int tid = threadIdx.x;
  if (tid < 32) {
    int r = rowbase + tid;
    stok[tid] = (r < rowhi) ? tok_ids[r] : -1;
  }

  const float* Wwp = Ww + (size_t)e * DMODEL * DFFD + (size_t)ft * 128;
  const float* Wvp = Wv + (size_t)e * DMODEL * DFFD + (size_t)ft * 128;

  const int wave = tid >> 5, lane = tid & 31;
  const int n = lane & 15;
  const int koff = (lane < 16) ? 0 : 8;   // 16-bit A/B lane-half K offset
  v8f aW0 = {}, aW1 = {}, aV0 = {}, aV1 = {};

  for (int kc = 0; kc < DMODEL; kc += 32) {
    __syncthreads();
    if (kc + 32 < DMODEL) {   // stream next weight chunk (L2-resident)
      __builtin_prefetch(&Wwp[(size_t)(kc + 32 + (tid >> 4)) * DFFD + (tid & 15) * 8], 0, 1);
      __builtin_prefetch(&Wvp[(size_t)(kc + 32 + (tid >> 4)) * DFFD + (tid & 15) * 8], 0, 1);
    }
    for (int idx = tid; idx < 32 * 32; idx += 256) {   // A chunk 32 rows x 32 k
      int r = idx >> 5, k = idx & 31;
      int tok = stok[r];
      xT[r * KSTR + k] = (tok >= 0) ? f2bf(x[(size_t)tok * DMODEL + kc + k])
                                    : (unsigned short)0;
    }
    for (int idx = tid; idx < 32 * 128; idx += 256) {  // B chunks, transposed
      int kk = idx >> 7, ff = idx & 127;
      wwT[ff * WSTR + kk] = f2bf(Wwp[(size_t)(kc + kk) * DFFD + ff]);
      wvT[ff * WSTR + kk] = f2bf(Wvp[(size_t)(kc + kk) * DFFD + ff]);
    }
    __syncthreads();
    FragB a0, a1, bwf, bvf;
#pragma unroll
    for (int v = 0; v < 4; ++v) {
      int kl = 2 * v + koff;
      a0.u[v]      = *(const unsigned*)&xT[n * KSTR + kl];
      a0.u[v + 4]  = *(const unsigned*)&xT[n * KSTR + kl + 16];
      a1.u[v]      = *(const unsigned*)&xT[(16 + n) * KSTR + kl];
      a1.u[v + 4]  = *(const unsigned*)&xT[(16 + n) * KSTR + kl + 16];
      bwf.u[v]     = *(const unsigned*)&wwT[(wave * 16 + n) * WSTR + kl];
      bwf.u[v + 4] = *(const unsigned*)&wwT[(wave * 16 + n) * WSTR + kl + 16];
      bvf.u[v]     = *(const unsigned*)&wvT[(wave * 16 + n) * WSTR + kl];
      bvf.u[v + 4] = *(const unsigned*)&wvT[(wave * 16 + n) * WSTR + kl + 16];
    }
    aW0 = __builtin_amdgcn_wmma_f32_16x16x32_bf16(false, a0.v, false, bwf.v,
                                                  (short)0, aW0, false, false);
    aV0 = __builtin_amdgcn_wmma_f32_16x16x32_bf16(false, a0.v, false, bvf.v,
                                                  (short)0, aV0, false, false);
    aW1 = __builtin_amdgcn_wmma_f32_16x16x32_bf16(false, a1.v, false, bwf.v,
                                                  (short)0, aW1, false, false);
    aV1 = __builtin_amdgcn_wmma_f32_16x16x32_bf16(false, a1.v, false, bvf.v,
                                                  (short)0, aV1, false, false);
  }

  const int fg = ft * 128 + wave * 16 + n;
  const float bwv = bw[e * DFFD + fg], bvv = bv[e * DFFD + fg];
  const int mbase = (lane < 16) ? 0 : 8;
#pragma unroll
  for (int i = 0; i < 8; ++i) {
    int m = i + mbase;
    int r0 = rowbase + m;
    if (r0 < rowhi) {
      float wx = aW0[i] + bwv;
      float vx = aV0[i] + bvv;
      h[(size_t)r0 * DFFD + fg] = f2bf((wx / (1.f + __expf(-wx))) * vx);
    }
    int r1 = rowbase + 16 + m;
    if (r1 < rowhi) {
      float wx = aW1[i] + bwv;
      float vx = aV1[i] + bvv;
      h[(size_t)r1 * DFFD + fg] = f2bf((wx / (1.f + __expf(-wx))) * vx);
    }
  }
}

// --------------------- expert down-projection + combine --------------------
// Block tile: 32 rows x 128 D cols. A-tile (bf16 h) staged via async-to-LDS
// B128 copies (ASYNCcnt) when the builtin exists; 16B-aligned 80B LDS rows.
#define HSTR 40

__global__ __launch_bounds__(256) void moe_expert_down(
    const unsigned short* __restrict__ h, const float* __restrict__ Wo,
    const float* __restrict__ bo, const int* __restrict__ tok_ids,
    const float* __restrict__ tok_w, const int* __restrict__ offsets,
    float* __restrict__ outp) {
  __shared__ unsigned short hT[32 * HSTR];
  __shared__ unsigned short woT[128 * WSTR];

  const int dt = blockIdx.x;             // D/128 tiles
  const int mt = blockIdx.y;
  const int e  = blockIdx.z;
  const int rowlo = offsets[e], rowhi = offsets[e + 1];
  const int rowbase = rowlo + mt * 32;
  if (rowbase >= rowhi) return;

  const int tid = threadIdx.x;
  // Pre-zero A tile so EXEC-masked (out-of-range) rows stay deterministic 0.
  for (int idx = tid; idx < 32 * HSTR; idx += 256) hT[idx] = 0;

  const float* Wop = Wo + (size_t)e * DFFD * DMODEL + (size_t)dt * 128;
  const int wave = tid >> 5, lane = tid & 31;
  const int n = lane & 15;
  const int koff = (lane < 16) ? 0 : 8;
  v8f acc0 = {}, acc1 = {};

  for (int kc = 0; kc < DFFD; kc += 32) {
    __syncthreads();
    if (tid < 128) {   // 32 rows x 64B: one B128 copy per lane
      int r = tid >> 2, kp = tid & 3;
      int rr = rowbase + r;
      if (rr < rowhi) {
#if HAVE_ASYNC_LDS
        __builtin_amdgcn_global_load_async_to_lds_b128(
            (AS1 v4i*)(h + (size_t)rr * DFFD + kc + kp * 8),
            (AS3 v4i*)&hT[r * HSTR + kp * 8], 0, 0);
#else
        const uint4 val = *(const uint4*)(h + (size_t)rr * DFFD + kc + kp * 8);
        *(uint4*)&hT[r * HSTR + kp * 8] = val;
#endif
      }
    }
    if (kc + 32 < DFFD)
      __builtin_prefetch(&Wop[(size_t)(kc + 32 + (tid >> 4)) * DMODEL + (tid & 15) * 8], 0, 1);
    for (int idx = tid; idx < 32 * 128; idx += 256) {
      int kk = idx >> 7, ff = idx & 127;
      woT[ff * WSTR + kk] = f2bf(Wop[(size_t)(kc + kk) * DMODEL + ff]);
    }
    wait_async_lds();
    __syncthreads();
    FragB a0, a1, b;
#pragma unroll
    for (int v = 0; v < 4; ++v) {
      int kl = 2 * v + koff;
      a0.u[v]     = *(const unsigned*)&hT[n * HSTR + kl];
      a0.u[v + 4] = *(const unsigned*)&hT[n * HSTR + kl + 16];
      a1.u[v]     = *(const unsigned*)&hT[(16 + n) * HSTR + kl];
      a1.u[v + 4] = *(const unsigned*)&hT[(16 + n) * HSTR + kl + 16];
      b.u[v]      = *(const unsigned*)&woT[(wave * 16 + n) * WSTR + kl];
      b.u[v + 4]  = *(const unsigned*)&woT[(wave * 16 + n) * WSTR + kl + 16];
    }
    acc0 = __builtin_amdgcn_wmma_f32_16x16x32_bf16(false, a0.v, false, b.v,
                                                   (short)0, acc0, false, false);
    acc1 = __builtin_amdgcn_wmma_f32_16x16x32_bf16(false, a1.v, false, b.v,
                                                   (short)0, acc1, false, false);
  }

  const int dg = dt * 128 + wave * 16 + n;
  const float bov = bo[e * DMODEL + dg];
  const int mbase = (lane < 16) ? 0 : 8;
#pragma unroll
  for (int i = 0; i < 8; ++i) {
    int m = i + mbase;
    int r0 = rowbase + m;
    if (r0 < rowhi)
      atomicAdd(&outp[(size_t)tok_ids[r0] * DMODEL + dg],
                tok_w[r0] * (acc0[i] + bov));
    int r1 = rowbase + 16 + m;
    if (r1 < rowhi)
      atomicAdd(&outp[(size_t)tok_ids[r1] * DMODEL + dg],
                tok_w[r1] * (acc1[i] + bov));
  }
}

// ------------------------------ launch -------------------------------------
extern "C" void kernel_launch(void* const* d_in, const int* in_sizes, int n_in,
                              void* d_out, int out_size, void* d_ws, size_t ws_size,
                              hipStream_t stream) {
  (void)in_sizes; (void)n_in; (void)out_size; (void)ws_size;
  const float* x  = (const float*)d_in[0];
  const float* Wg = (const float*)d_in[1];
  const float* bg = (const float*)d_in[2];
  const float* Ww = (const float*)d_in[3];
  const float* bw = (const float*)d_in[4];
  const float* Wv = (const float*)d_in[5];
  const float* bv = (const float*)d_in[6];
  const float* Wo = (const float*)d_in[7];
  const float* bo = (const float*)d_in[8];
  float* outp = (float*)d_out;                       // final, then logits

  char* ws = (char*)d_ws;
  constexpr size_t SEL_OFF  = 0;                               // 2N ints
  constexpr size_t SELW_OFF = SEL_OFF  + (size_t)2 * NTOK * 4; // 2N floats
  constexpr size_t CNT_OFF  = SELW_OFF + (size_t)2 * NTOK * 4; // E ints
  constexpr size_t OFFS_OFF = CNT_OFF  + 256;                  // E+1 ints
  constexpr size_t TID_OFF  = OFFS_OFF + 256;                  // 2N ints
  constexpr size_t TW_OFF   = TID_OFF  + (size_t)2 * NTOK * 4; // 2N floats
  constexpr size_t H_OFF    = TW_OFF   + (size_t)2 * NTOK * 4; // 2N x DFF bf16
  int*   sel     = (int*)(ws + SEL_OFF);
  float* selw    = (float*)(ws + SELW_OFF);
  int*   counts  = (int*)(ws + CNT_OFF);
  int*   offsets = (int*)(ws + OFFS_OFF);
  int*   tok_ids = (int*)(ws + TID_OFF);
  float* tok_w   = (float*)(ws + TW_OFF);
  unsigned short* hbuf = (unsigned short*)(ws + H_OFF);

  moe_zero<<<(NTOK * DMODEL + 255) / 256, 256, 0, stream>>>(outp, counts);
  moe_router<<<NTOK / 8, 256, 0, stream>>>(x, Wg, bg,
                                           outp + (size_t)NTOK * DMODEL,
                                           sel, selw, counts);
  moe_scan<<<1, 32, 0, stream>>>(counts, offsets);
  moe_place<<<NEXP, 32, 0, stream>>>(sel, selw, offsets, tok_ids, tok_w);
  moe_expert_up<<<dim3(DFFD / 128, NTOK / 32, NEXP), 256, 0, stream>>>(
      x, Ww, bw, Wv, bv, tok_ids, offsets, hbuf);
  moe_expert_down<<<dim3(DMODEL / 128, NTOK / 32, NEXP), 256, 0, stream>>>(
      hbuf, Wo, bo, tok_ids, tok_w, offsets, outp);
}
